// FraudAwareAugmentor_31413390803482
// MI455X (gfx1250) — matrix operand order repaired and checked by reading, not compiled, not run on hardware
//
#include <hip/hip_runtime.h>
#include <hip/hip_bf16.h>

// Problem constants (from the reference harness)
#define UU   4096     // users  (C is UU x UU)
#define II   20000    // items  (K dimension of C = A * A^T), multiple of 32
#define HID  16
#define NTOP 16
#define PITCH 20      // dwords per 32-bf16 LDS row: conflict-free AND 16B-aligned

typedef __attribute__((ext_vector_type(16))) __bf16 v16bf;
typedef __attribute__((ext_vector_type(8)))  float  v8f;

// ---------------------------------------------------------------- utilities
__global__ void zero_kernel(float* p, size_t n) {
    size_t i = (size_t)blockIdx.x * blockDim.x + threadIdx.x;
    size_t stride = (size_t)gridDim.x * blockDim.x;
    for (; i < n; i += stride) p[i] = 0.0f;
}

// -------------------------------------------------- step 1: degree counting
// user_idx == arange(U) in the harness, so user_map[g]=g and item_map[g]=g-U.
__global__ void degree_kernel(const int* __restrict__ ei, int E,
                              float* cnt_u, float* cnt_i) {
    int e = blockIdx.x * blockDim.x + threadIdx.x;
    if (e >= E) return;
    int s = ei[e], d = ei[E + e];
    if (s >= 0 && s < UU && d >= UU && d < UU + II) {
        atomicAdd(&cnt_u[s], 1.0f);
        atomicAdd(&cnt_i[d - UU], 1.0f);
    }
}

// d_u = log1p(count); Dh = sqrt(d_u); also emit d_u output
__global__ void user_post_kernel(const float* cnt_u, float* sqrtDu, float* du_out) {
    int u = blockIdx.x * blockDim.x + threadIdx.x;
    if (u >= UU) return;
    float d = log1pf(cnt_u[u]);
    sqrtDu[u] = sqrtf(d);
    du_out[u] = d;
}

// ---------------------------------------- step 2: item MLP -> sqrt(d_i)*w_i
__global__ void item_post_kernel(const float* cnt_i, const float* __restrict__ fraud,
                                 const float* __restrict__ W1, const float* __restrict__ b1,
                                 const float* __restrict__ W2, const float* __restrict__ b2,
                                 float* itemFactor) {
    int i = blockIdx.x * blockDim.x + threadIdx.x;
    if (i >= II) return;
    float d = log1pf(cnt_i[i]);
    float f = fraud[i];
    float s = b2[0];
#pragma unroll
    for (int j = 0; j < HID; ++j) {
        float h = fmaf(d, W1[j], fmaf(f, W1[HID + j], b1[j]));   // x @ W1 + b1
        h = fmaxf(h, 0.0f);                                       // relu
        s = fmaf(h, W2[j], s);                                    // h @ W2 + b2
    }
    float w = 1.0f / (1.0f + expf(-s));                           // sigmoid
    itemFactor[i] = sqrtf(d) * w;                                 // sqrt(d_i) * w_i
}

// ------------------------------- step 3: scatter-add dense A[U,I] (f32 acc)
__global__ void build_A_kernel(const int* __restrict__ ei, int E,
                               const float* __restrict__ sqrtDu,
                               const float* __restrict__ itemFactor,
                               float* A) {
    int e = blockIdx.x * blockDim.x + threadIdx.x;
    if (e >= E) return;
    int s = ei[e], d = ei[E + e];
    if (s >= 0 && s < UU && d >= UU && d < UU + II) {
        int it = d - UU;
        atomicAdd(&A[(size_t)s * II + it], sqrtDu[s] * itemFactor[it]);
    }
}

// f32 -> bf16 for the WMMA GEMM
__global__ void convert_kernel(const float* __restrict__ A, __hip_bfloat16* Abf, size_t n) {
    size_t i = (size_t)blockIdx.x * blockDim.x + threadIdx.x;
    size_t stride = (size_t)gridDim.x * blockDim.x;
    for (; i < n; i += stride) Abf[i] = __float2bfloat16(A[i]);
}

// ------------------------------------------ step 4: Cn = (A A^T) * Dh ⊗ Dh
// 128x128 C tile per 256-thread block (8 wave32s). Wave w owns 2 row-tiles
// x 4 col-tiles => 8 WMMAs per K=32 step from 6 fragment loads. Double-
// buffered LDS (one barrier per step), global loads staged 2 slabs ahead.
// Triangular grid (C symmetric), mirrored store for off-diagonal blocks.
__global__ void __launch_bounds__(256)
gemm_wmma_kernel(const unsigned short* __restrict__ Abf,
                 const float* __restrict__ sqrtDu,
                 float* __restrict__ C) {
    int bx = blockIdx.x, by = blockIdx.y;
    if (bx > by) return;                        // symmetry: lower triangle only
    int rowBase = by * 128;
    int colBase = bx * 128;

    // [buf][A/B][128 rows * PITCH dwords]  = 2 * 2 * 10240 B = 40 KB
    __shared__ unsigned int sh[2][2][128 * PITCH];

    int t    = threadIdx.x;
    int wave = t >> 5;
    int lane = t & 31;
    int hl   = lane & 15;                       // half-lane id
    int hi   = lane >> 4;                       // lane half (0/1)
    int wr   = (wave & 3) * 2;                  // first of 2 row subtiles
    int wc   = (wave >> 2) * 4;                 // first of 4 col subtiles

    v8f acc[2][4];
#pragma unroll
    for (int i = 0; i < 2; ++i)
#pragma unroll
        for (int j = 0; j < 4; ++j) acc[i][j] = (v8f){0.f,0.f,0.f,0.f,0.f,0.f,0.f,0.f};

    int ldRow   = t >> 2;                       // 0..63 (also loads ldRow+64)
    int ldChunk = t & 3;                        // uint4 chunk within 32-bf16 row

    const size_t aOff0 = (size_t)(rowBase + ldRow)      * II + (size_t)ldChunk * 8;
    const size_t aOff1 = (size_t)(rowBase + ldRow + 64) * II + (size_t)ldChunk * 8;
    const size_t bOff0 = (size_t)(colBase + ldRow)      * II + (size_t)ldChunk * 8;
    const size_t bOff1 = (size_t)(colBase + ldRow + 64) * II + (size_t)ldChunk * 8;
    const int stA0 = ldRow * PITCH + ldChunk * 4;        // 16B-aligned LDS slots
    const int stA1 = (ldRow + 64) * PITCH + ldChunk * 4;

    union Frag { uint4 q[2]; v16bf v; };
    const int steps = II / 32;                  // 625

    // ---- prologue: slab 0 -> buf0; stage slab 1 in registers
    uint4 sA0 = *(const uint4*)(Abf + aOff0);
    uint4 sA1 = *(const uint4*)(Abf + aOff1);
    uint4 sB0 = *(const uint4*)(Abf + bOff0);
    uint4 sB1 = *(const uint4*)(Abf + bOff1);
    *(uint4*)&sh[0][0][stA0] = sA0;
    *(uint4*)&sh[0][0][stA1] = sA1;
    *(uint4*)&sh[0][1][stA0] = sB0;
    *(uint4*)&sh[0][1][stA1] = sB1;
    sA0 = *(const uint4*)(Abf + aOff0 + 32);
    sA1 = *(const uint4*)(Abf + aOff1 + 32);
    sB0 = *(const uint4*)(Abf + bOff0 + 32);
    sB1 = *(const uint4*)(Abf + bOff1 + 32);
    __syncthreads();

    for (int s = 0; s < steps; ++s) {
        const int cur = s & 1;
        const unsigned int* shA = &sh[cur][0][0];
        const unsigned int* shB = &sh[cur][1][0];

        // A fragments (16x32 bf16): lane-half hi -> K groups {hi*8.., 16+hi*8..}
        Frag fa[2], fb[4];
#pragma unroll
        for (int i = 0; i < 2; ++i) {
            const uint4* pA = (const uint4*)&shA[((wr + i) * 16 + hl) * PITCH + hi * 4];
            fa[i].q[0] = pA[0];                 // dwords +0..3
            fa[i].q[1] = pA[2];                 // dwords +8..11
        }
        // B fragments (32x16 bf16): col N = hl, K groups {hi*8.., 16+hi*8..}
#pragma unroll
        for (int j = 0; j < 4; ++j) {
            const uint4* pB = (const uint4*)&shB[((wc + j) * 16 + hl) * PITCH + hi * 4];
            fb[j].q[0] = pB[0];
            fb[j].q[1] = pB[2];
        }

        // store staged slab s+1 into the other buffer (safe: nobody reads it)
        if (s + 1 < steps) {
            unsigned int* dA = &sh[cur ^ 1][0][0];
            unsigned int* dB = &sh[cur ^ 1][1][0];
            *(uint4*)&dA[stA0] = sA0;
            *(uint4*)&dA[stA1] = sA1;
            *(uint4*)&dB[stA0] = sB0;
            *(uint4*)&dB[stA1] = sB1;
        }
        // kick off global loads for slab s+2
        if (s + 2 < steps) {
            const size_t kb2 = (size_t)(s + 2) * 32;
            sA0 = *(const uint4*)(Abf + aOff0 + kb2);
            sA1 = *(const uint4*)(Abf + aOff1 + kb2);
            sB0 = *(const uint4*)(Abf + bOff0 + kb2);
            sB1 = *(const uint4*)(Abf + bOff1 + kb2);
            if (s + 3 < steps) {                // global_prefetch_b8 of next slab
                __builtin_prefetch(Abf + aOff0 + kb2 + 32, 0, 3);
                __builtin_prefetch(Abf + bOff0 + kb2 + 32, 0, 3);
            }
        }

        // 8 WMMAs; independent accumulators hide the bf16 WMMA->WMMA hazard
#pragma unroll
        for (int i = 0; i < 2; ++i)
#pragma unroll
            for (int j = 0; j < 4; ++j)
                acc[i][j] = __builtin_amdgcn_wmma_f32_16x16x32_bf16(
                    false, fa[i].v, false, fb[j].v, (short)0, acc[i][j], false, false);

        __syncthreads();                        // buf[cur] reads & buf[cur^1] writes done
    }

    // Epilogue: fuse Cn = C * Dh[row]*Dh[col]; mirror across the diagonal.
#pragma unroll
    for (int i = 0; i < 2; ++i) {
#pragma unroll
        for (int j = 0; j < 4; ++j) {
            int colT = colBase + (wc + j) * 16 + hl;
            float dhc = sqrtDu[colT];
#pragma unroll
            for (int r = 0; r < 8; ++r) {
                int row = rowBase + (wr + i) * 16 + hi * 8 + r;   // C/D VGPR layout
                float v = acc[i][j][r] * sqrtDu[row] * dhc;
                C[(size_t)row * UU + colT] = v;
                if (bx != by) C[(size_t)colT * UU + row] = v;     // L2-resident mirror
            }
        }
    }
}

// ------------------- step 5+6: per-row top-16 and symmetric scatter into S
__global__ void __launch_bounds__(256)
topk_scatter_kernel(const float* __restrict__ C, float* S) {
    int row = blockIdx.x;
    __shared__ float vals[UU];                 // 16 KB row cache
    __shared__ float redV[256];
    __shared__ int   redI[256];
    int t = threadIdx.x;
    for (int j = t; j < UU; j += 256) vals[j] = C[(size_t)row * UU + j];
    __syncthreads();

    for (int it = 0; it < NTOP; ++it) {
        float lv = -1.0f; int li = 0x7fffffff;
        for (int j = t; j < UU; j += 256) {    // Cn >= 0, sentinel below is -2
            float v = vals[j];
            if (v > lv || (v == lv && j < li)) { lv = v; li = j; }
        }
        redV[t] = lv; redI[t] = li;
        __syncthreads();
        for (int s = 128; s > 0; s >>= 1) {
            if (t < s) {
                float ov = redV[t + s]; int oi = redI[t + s];
                if (ov > redV[t] || (ov == redV[t] && oi < redI[t])) { redV[t] = ov; redI[t] = oi; }
            }
            __syncthreads();
        }
        if (t == 0) {
            float v = redV[0]; int c = redI[0];
            atomicAdd(&S[(size_t)row * UU + c], 0.5f * v);   // T
            atomicAdd(&S[(size_t)c * UU + row], 0.5f * v);   // T^T (dedup by summation)
            vals[c] = -2.0f;                                  // mask out
        }
        __syncthreads();
    }
}

// --------------------------------------------------------------- launcher
extern "C" void kernel_launch(void* const* d_in, const int* in_sizes, int n_in,
                              void* d_out, int out_size, void* d_ws, size_t ws_size,
                              hipStream_t stream) {
    const float* fraud = (const float*)d_in[0];
    const float* W1    = (const float*)d_in[1];
    const float* b1    = (const float*)d_in[2];
    const float* W2    = (const float*)d_in[3];
    const float* b2    = (const float*)d_in[4];
    const int*   ei    = (const int*)d_in[5];
    const int    E     = in_sizes[5] / 2;

    float* out    = (float*)d_out;
    float* S      = out;                          // [U*U]
    float* du_out = out + (size_t)UU * UU;        // [U]

    // Workspace layout (~533 MB): cnt_u|cnt_i|sqrtDu|itemFactor|A_f32|A_bf16|C
    float* ws         = (float*)d_ws;
    float* cnt_u      = ws;                       // 4096   (cnt_u & cnt_i contiguous)
    float* cnt_i      = cnt_u + UU;               // 20000
    float* sqrtDu     = cnt_i + II;               // 4096
    float* itemFactor = sqrtDu + UU;              // 20000
    float* Af         = itemFactor + II;          // U*I f32   (327.7 MB)
    __hip_bfloat16* Abf = (__hip_bfloat16*)(Af + (size_t)UU * II);    // U*I bf16 (163.8 MB)
    float* C = (float*)((char*)Abf + (size_t)UU * II * sizeof(__hip_bfloat16)); // U*U f32

    // zero accumulated buffers (fresh every call: deterministic replay)
    zero_kernel<<<256, 256, 0, stream>>>(cnt_u, (size_t)(UU + II));
    zero_kernel<<<4096, 256, 0, stream>>>(Af, (size_t)UU * II);
    zero_kernel<<<4096, 256, 0, stream>>>(S, (size_t)UU * UU);

    degree_kernel<<<(E + 255) / 256, 256, 0, stream>>>(ei, E, cnt_u, cnt_i);
    user_post_kernel<<<UU / 256, 256, 0, stream>>>(cnt_u, sqrtDu, du_out);
    item_post_kernel<<<(II + 255) / 256, 256, 0, stream>>>(cnt_i, fraud, W1, b1, W2, b2, itemFactor);
    build_A_kernel<<<(E + 255) / 256, 256, 0, stream>>>(ei, E, sqrtDu, itemFactor, Af);
    convert_kernel<<<4096, 256, 0, stream>>>(Af, Abf, (size_t)UU * II);

    gemm_wmma_kernel<<<dim3(UU / 128, UU / 128), 256, 0, stream>>>(
        (const unsigned short*)Abf, sqrtDu, C);

    topk_scatter_kernel<<<UU, 256, 0, stream>>>(C, S);
}
